// WaterLevelModel_9603546874319
// MI455X (gfx1250) — matrix operand order repaired and checked by reading, not compiled, not run on hardware
//
#include <hip/hip_runtime.h>
#include <stdint.h>

// ---------------------------------------------------------------------------
// WaterLevelModel for MI455X (gfx1250, wave32).
//
//  * Edge-weight matrices W_e = MLP(edge_attr) computed ONCE per launch with
//    v_wmma_f32_16x16x32_f16 (K=128 GEMM -> 4 chained WMMAs / 16x16 tile).
//  * W_e stored in f16, transposed [e][out][in]: 163.8 MB -> resident in the
//    192 MB L2, so the 64 recurrent steps stream it from L2 not HBM (f32
//    would be 327 MB and cost ~21 GB of HBM traffic over the scan).
//  * MLP epilogue staged through LDS so W_e is written with coalesced b128
//    stores instead of 16x-amplified scattered b16 stores.
//  * Per-edge 1x16 x 16x16 matvec: distinct matrix per edge -> VALU f32;
//    hc row broadcast via same-address float4 loads (L1 broadcast), scatter
//    -mean via global_atomic_add_f32 at L2.
//  * GRU / root-conv / head: per-node VALU, wave-uniform weights -> s_loads.
// Workspace: ~E*512B (W_e f16) + ~4MB state buffers (~170 MB).
// ---------------------------------------------------------------------------

typedef __attribute__((ext_vector_type(16))) _Float16 v16h;
typedef __attribute__((ext_vector_type(8)))  _Float16 v8h;
typedef __attribute__((ext_vector_type(8)))  float    v8f;

#define Hd 16
#define Tn 64

__device__ __forceinline__ float sigmoid_f(float x) {
  return 1.0f / (1.0f + __expf(-x));
}
__device__ __forceinline__ float tanh_f(float x) {
  return 2.0f / (1.0f + __expf(-2.0f * x)) - 1.0f;
}

// ---------------------------------------------------------------------------
// Setup kernels
// ---------------------------------------------------------------------------

__global__ void k_init(float* __restrict__ hbuf, float* __restrict__ countf,
                       int BN) {
  int i = blockIdx.x * blockDim.x + threadIdx.x;
  if (i < BN * Hd) hbuf[i] = 0.0f;
  if (i < BN) countf[i] = 0.0f;
}

__global__ void k_count(const int* __restrict__ eidx, float* __restrict__ countf,
                        int E) {
  int e = blockIdx.x * blockDim.x + threadIdx.x;
  if (e < E) atomicAdd(&countf[eidx[E + e]], 1.0f);
}

__global__ void k_invden(const float* __restrict__ countf,
                         float* __restrict__ invden, int BN) {
  int n = blockIdx.x * blockDim.x + threadIdx.x;
  if (n < BN) {
    float c = countf[n];
    invden[n] = 1.0f / (c > 0.0f ? c : 1.0f);
  }
}

// Repack mlp_W2 [128,256] f32 into lane-ordered f16 B fragments for
// v_wmma_f32_16x16x32_f16.  Fragment (nt,kc): lane&15 = column N within the
// N-tile, lane<16 holds K {0..7,16..23}, lane>=16 {8..15,24..31} of the chunk.
__global__ void k_repackB(const float* __restrict__ W2,
                          _Float16* __restrict__ Bfrag) {
  int idx = blockIdx.x * blockDim.x + threadIdx.x;  // 16*4*32*16 = 32768
  if (idx < 16 * 4 * 32 * 16) {
    int h    = idx & 15;
    int lane = (idx >> 4) & 31;
    int kc   = (idx >> 9) & 3;
    int nt   = idx >> 11;
    int K = kc * 32 + ((h < 8) ? ((lane >> 4) * 8 + h) : (16 + (lane >> 4) * 8 + (h - 8)));
    int N = nt * 16 + (lane & 15);
    Bfrag[idx] = (_Float16)W2[K * 256 + N];
  }
}

// ---------------------------------------------------------------------------
// W_e = relu(nan0(ea) @ W1 + b1) @ W2 + b2, one 16-edge M-tile per wave.
// Layer 1 (K=4) in VALU straight into the WMMA A-fragment layout; layer 2
// (K=128) as 4 chained v_wmma_f32_16x16x32_f16 per 16-wide N-tile.
// Output staged in LDS, then stored transposed f16 W_T[e*256 + o*16 + i]
// with coalesced 16B/lane stores.
// ---------------------------------------------------------------------------
__global__ void k_mlp_we(const float* __restrict__ eattr,
                         const float* __restrict__ W1,
                         const float* __restrict__ b1,
                         const float* __restrict__ b2,
                         const _Float16* __restrict__ Bfrag,
                         _Float16* __restrict__ W_T,
                         int E, int numTiles) {
  __shared__ float sW1[4 * 128];
  __shared__ float sb1[128];
  __shared__ float sb2[256];
  __shared__ _Float16 sOut[4][16 * 256];  // 8KB per wave
  for (int i = threadIdx.x; i < 512; i += blockDim.x) sW1[i] = W1[i];
  for (int i = threadIdx.x; i < 128; i += blockDim.x) sb1[i] = b1[i];
  for (int i = threadIdx.x; i < 256; i += blockDim.x) sb2[i] = b2[i];
  __syncthreads();

  int wave = threadIdx.x >> 5;
  int lane = threadIdx.x & 31;
  int tile = blockIdx.x * (blockDim.x >> 5) + wave;
  bool active = (tile < numTiles);  // wave-uniform

  int m  = lane & 15;   // A row (edge within tile) == D column (out dim)
  int hi = lane >> 4;
  _Float16* so = &sOut[wave][0];

  if (active) {
    long e  = (long)tile * 16 + m;
    long ec = (e < E) ? e : (long)(E - 1);

    float ea[4];
#pragma unroll
    for (int f = 0; f < 4; ++f) {
      float v = eattr[ec * 4 + f];
      ea[f] = (v == v) ? v : 0.0f;  // nan_to_num
    }

    // A fragments: lane<16 holds K {0..7,16..23}, lane>=16 {8..15,24..31}
    v16h A[4];
#pragma unroll
    for (int kc = 0; kc < 4; ++kc) {
#pragma unroll
      for (int h = 0; h < 16; ++h) {
        int krel = (h < 8) ? (hi * 8 + h) : (16 + hi * 8 + (h - 8));
        int k = kc * 32 + krel;
        float v = sb1[k] + ea[0] * sW1[k] + ea[1] * sW1[128 + k] +
                  ea[2] * sW1[256 + k] + ea[3] * sW1[384 + k];
        A[kc][h] = (_Float16)(v > 0.0f ? v : 0.0f);  // relu
      }
    }

    const v16h* Bf = (const v16h*)Bfrag;
    for (int nt = 0; nt < 16; ++nt) {  // N-tile nt == input row i of W_e
      v8f c = {};
#pragma unroll
      for (int kc = 0; kc < 4; ++kc) {
        c = __builtin_amdgcn_wmma_f32_16x16x32_f16(
            false, A[kc], false, Bf[(nt * 4 + kc) * 32 + lane],
            (short)0, c, false, false);
      }
      float b2v = sb2[nt * 16 + m];
#pragma unroll
      for (int v = 0; v < 8; ++v) {  // D row = v + 8*hi
        so[(v + 8 * hi) * 256 + m * 16 + nt] = (_Float16)(c[v] + b2v);
      }
    }
  }

  // Same-wave LDS store->load is in-order in HW (DScnt); just stop the
  // compiler from reordering across the exchange.
  __builtin_amdgcn_wave_barrier();

  if (active) {
    const v8h* sp = (const v8h*)so;
    v8h* gp = (v8h*)W_T;
    long base = (long)tile * 512;  // 512 v8h units per 16-edge tile
    bool fullTile = ((long)(tile + 1) * 16 <= (long)E);  // wave-uniform
    if (fullTile) {
#pragma unroll
      for (int c2 = 0; c2 < 16; ++c2) {
        gp[base + c2 * 32 + lane] = sp[c2 * 32 + lane];  // coalesced 512B/row
      }
    } else {
#pragma unroll
      for (int c2 = 0; c2 < 16; ++c2) {
        int u = c2 * 32 + lane;
        long erow = (long)tile * 16 + (u >> 5);
        if (erow < E) gp[base + u] = sp[u];
      }
    }
  }
}

// ---------------------------------------------------------------------------
// Per-step kernels
// ---------------------------------------------------------------------------

// GRUCell (gate order r,z,n) for one node per lane; also zeroes agg for the
// upcoming scatter. Weight accesses are wave-uniform -> scalar loads.
__global__ void k_gru(const float* __restrict__ runoff,
                      const float* __restrict__ Wih,
                      const float* __restrict__ Whh,
                      const float* __restrict__ bih,
                      const float* __restrict__ bhh,
                      const float* __restrict__ hbuf,
                      float* __restrict__ hc,
                      float* __restrict__ agg,
                      int BN, int t) {
  int n = blockIdx.x * blockDim.x + threadIdx.x;
  if (n >= BN) return;
  float h[Hd];
#pragma unroll
  for (int i = 0; i < Hd; ++i) h[i] = hbuf[(long)n * Hd + i];
  float x = runoff[(long)n * Tn + t];

  float r[Hd], z[Hd], hcv[Hd];
#pragma unroll
  for (int j = 0; j < Hd; ++j) {
    float g = Wih[j] * x + bih[j] + bhh[j];
#pragma unroll
    for (int i = 0; i < Hd; ++i) g += Whh[j * Hd + i] * h[i];
    r[j] = sigmoid_f(g);
  }
#pragma unroll
  for (int j = 0; j < Hd; ++j) {
    int jj = Hd + j;
    float g = Wih[jj] * x + bih[jj] + bhh[jj];
#pragma unroll
    for (int i = 0; i < Hd; ++i) g += Whh[jj * Hd + i] * h[i];
    z[j] = sigmoid_f(g);
  }
#pragma unroll
  for (int j = 0; j < Hd; ++j) {
    int jj = 2 * Hd + j;
    float gx = Wih[jj] * x + bih[jj];
    float gh = bhh[jj];
#pragma unroll
    for (int i = 0; i < Hd; ++i) gh += Whh[jj * Hd + i] * h[i];
    float ng = tanh_f(gx + r[j] * gh);
    hcv[j] = (1.0f - z[j]) * ng + z[j] * h[j];
  }
#pragma unroll
  for (int j = 0; j < Hd; ++j) {
    hc[(long)n * Hd + j] = hcv[j];
    agg[(long)n * Hd + j] = 0.0f;
  }
}

// msg[e] = hc[src[e]] @ W_e[e]; 16 lanes per edge, lane j = output column.
// W_T gives each lane a contiguous 32 B f16 line; the hc row is fetched with
// four same-address float4 loads per group (L1 merges + broadcasts), which is
// cheaper than a 16-deep shuffle chain. Scatter via f32 global atomics at L2.
__global__ void k_msg(const int* __restrict__ eidx,
                      const float* __restrict__ hc,
                      const _Float16* __restrict__ W_T,
                      float* __restrict__ agg, int E) {
  int gid = blockIdx.x * blockDim.x + threadIdx.x;
  int e = gid >> 4;
  int j = gid & 15;
  if (e >= E) return;
  int s = eidx[e];
  int d = eidx[E + e];
  const float4* hp = (const float4*)(hc + (long)s * Hd);
  float4 h0 = hp[0], h1 = hp[1], h2 = hp[2], h3 = hp[3];
  const v8h* w = (const v8h*)(W_T + (long)e * 256 + j * 16);
  v8h w0 = w[0], w1 = w[1];
  float acc;
  acc  = h0.x * (float)w0[0];
  acc += h0.y * (float)w0[1];
  acc += h0.z * (float)w0[2];
  acc += h0.w * (float)w0[3];
  acc += h1.x * (float)w0[4];
  acc += h1.y * (float)w0[5];
  acc += h1.z * (float)w0[6];
  acc += h1.w * (float)w0[7];
  acc += h2.x * (float)w1[0];
  acc += h2.y * (float)w1[1];
  acc += h2.z * (float)w1[2];
  acc += h2.w * (float)w1[3];
  acc += h3.x * (float)w1[4];
  acc += h3.y * (float)w1[5];
  acc += h3.z * (float)w1[6];
  acc += h3.w * (float)w1[7];
  atomicAdd(&agg[(long)d * Hd + j], acc);
}

// h_new = agg/denom + hc @ conv_root + bias; out = h_new @ out_W + out_b
__global__ void k_fin(const float* __restrict__ hc,
                      const float* __restrict__ agg,
                      const float* __restrict__ invden,
                      const float* __restrict__ conv_root,
                      const float* __restrict__ conv_bias,
                      const float* __restrict__ out_W,
                      const float* __restrict__ out_b,
                      float* __restrict__ hbuf,
                      float* __restrict__ preds,
                      int BN, int t) {
  int n = blockIdx.x * blockDim.x + threadIdx.x;
  if (n >= BN) return;
  float inv = invden[n];
  float hcv[Hd];
#pragma unroll
  for (int k = 0; k < Hd; ++k) hcv[k] = hc[(long)n * Hd + k];
  float oacc = out_b[0];
#pragma unroll
  for (int i = 0; i < Hd; ++i) {
    float a = agg[(long)n * Hd + i] * inv + conv_bias[i];
#pragma unroll
    for (int k = 0; k < Hd; ++k) a += hcv[k] * conv_root[k * Hd + i];
    hbuf[(long)n * Hd + i] = a;
    oacc += a * out_W[i];
  }
  preds[(long)t * BN + n] = oacc;
}

// ---------------------------------------------------------------------------
// Launch
// ---------------------------------------------------------------------------
extern "C" void kernel_launch(void* const* d_in, const int* in_sizes, int n_in,
                              void* d_out, int out_size, void* d_ws,
                              size_t ws_size, hipStream_t stream) {
  const float* runoff    = (const float*)d_in[0];
  const int*   eidx      = (const int*)d_in[1];
  const float* eattr     = (const float*)d_in[2];
  const float* Wih       = (const float*)d_in[4];
  const float* Whh       = (const float*)d_in[5];
  const float* bih       = (const float*)d_in[6];
  const float* bhh       = (const float*)d_in[7];
  const float* W1        = (const float*)d_in[8];
  const float* b1        = (const float*)d_in[9];
  const float* W2        = (const float*)d_in[10];
  const float* b2        = (const float*)d_in[11];
  const float* conv_root = (const float*)d_in[12];
  const float* conv_bias = (const float*)d_in[13];
  const float* out_W     = (const float*)d_in[14];
  const float* out_b     = (const float*)d_in[15];
  float* preds = (float*)d_out;

  const int BN = in_sizes[0] / Tn;   // B==1 in this problem
  const int E  = in_sizes[1] / 2;

  // bump allocator over d_ws (256B aligned)
  uint8_t* p = (uint8_t*)d_ws;
  auto alloc = [&](size_t bytes) -> void* {
    void* r = (void*)p;
    p += (bytes + 255) & ~(size_t)255;
    return r;
  };
  _Float16* W_T    = (_Float16*)alloc((size_t)E * 256 * sizeof(_Float16));
  float*    hbuf   = (float*)alloc((size_t)BN * Hd * sizeof(float));
  float*    hcbuf  = (float*)alloc((size_t)BN * Hd * sizeof(float));
  float*    agg    = (float*)alloc((size_t)BN * Hd * sizeof(float));
  float*    countf = (float*)alloc((size_t)BN * sizeof(float));
  float*    invden = (float*)alloc((size_t)BN * sizeof(float));
  _Float16* Bfrag  = (_Float16*)alloc(32768 * sizeof(_Float16));
  (void)ws_size; (void)n_in; (void)out_size;

  k_init<<<(BN * Hd + 255) / 256, 256, 0, stream>>>(hbuf, countf, BN);
  k_count<<<(E + 255) / 256, 256, 0, stream>>>(eidx, countf, E);
  k_invden<<<(BN + 255) / 256, 256, 0, stream>>>(countf, invden, BN);
  k_repackB<<<(32768 + 255) / 256, 256, 0, stream>>>(W2, Bfrag);

  const int numTiles = (E + 15) / 16;
  k_mlp_we<<<(numTiles + 3) / 4, 128, 0, stream>>>(eattr, W1, b1, b2, Bfrag,
                                                   W_T, E, numTiles);

  const long msgThreads = (long)E * 16;
  for (int t = 0; t < Tn; ++t) {
    k_gru<<<(BN + 255) / 256, 256, 0, stream>>>(runoff, Wih, Whh, bih, bhh,
                                                hbuf, hcbuf, agg, BN, t);
    k_msg<<<(int)((msgThreads + 255) / 256), 256, 0, stream>>>(eidx, hcbuf,
                                                               W_T, agg, E);
    k_fin<<<(BN + 255) / 256, 256, 0, stream>>>(hcbuf, agg, invden, conv_root,
                                                conv_bias, out_W, out_b, hbuf,
                                                preds, BN, t);
  }
}
